// SequenceModel_76218489635479
// MI455X (gfx1250) — matrix operand order repaired and checked by reading, not compiled, not run on hardware
//
#include <hip/hip_runtime.h>
#include <hip/hip_bf16.h>
#include <math.h>

// ---------------------------------------------------------------------------
// GRU sequence model for MI455X (gfx1250, wave32, WMMA bf16).
//   Stage 1: swizzle Wi / [Wh_rz|Wh_n] fp32 -> bf16 in WMMA B-fragment order.
//   Stage 2: xg = X @ Wi + bi. One block per 16-row M tile; A converted to
//            bf16 once per block via LDS staging (X read exactly once = HBM
//            roofline), B fragments L2-resident, v_wmma_f32_16x16x32_bf16.
//   Stage 3: per-batch-slice scan: 8 WGs x 16 waves, each WG owns 16 batch
//            rows for all 512 steps; no inter-WG synchronization at all.
// ---------------------------------------------------------------------------

typedef __attribute__((ext_vector_type(16))) __bf16 bf16x16;
typedef __attribute__((ext_vector_type(8)))  float  f32x8;

union BFrag {
  bf16x16      v;     // 32 bytes = 8 VGPRs (WMMA 16-bit A or B fragment)
  uint4        q[2];
  unsigned int w[8];
};

__device__ __forceinline__ unsigned short f2bfu(float f) {
  unsigned int u = __float_as_uint(f);
  u += 0x7FFFu + ((u >> 16) & 1u);          // round-to-nearest-even
  return (unsigned short)(u >> 16);
}
__device__ __forceinline__ unsigned int packbf(float lo, float hi) {
  return (unsigned int)f2bfu(lo) | ((unsigned int)f2bfu(hi) << 16);
}

__device__ __forceinline__ f32x8 wmma_bf16(const BFrag& a, const BFrag& b, f32x8 c) {
  // v_wmma_f32_16x16x32_bf16  D = A(16x32) x B(32x16) + C(16x16 f32)
  return __builtin_amdgcn_wmma_f32_16x16x32_bf16(
      /*neg_a=*/false, a.v, /*neg_b=*/false, b.v,
      /*c_mod=*/(short)0, c, /*reuse_a=*/false, /*reuse_b=*/false);
}

static constexpr int T_STEPS = 512;
static constexpr int BATCH   = 128;
static constexpr int F_IN    = 1030;          // 32*32 + 6
static constexpr int H       = 512;
static constexpr int G3      = 3 * H;         // 1536
static constexpr int KC_X    = 33;            // ceil(1030/32) K-chunks, input GEMM
static constexpr int KC_H    = 16;            // 512/32 K-chunks, hidden GEMM
static constexpr int NT      = 96;            // 1536/16 N tiles
static constexpr int M_TILES = (T_STEPS * BATCH) / 16;   // 4096

static constexpr size_t XG_BYTES  = (size_t)T_STEPS * BATCH * G3 * 4;       // 402,653,184
static constexpr size_t WIB_BYTES = (size_t)KC_X * NT * 32 * 32;            //   3,244,032

// ---------------------------------------------------------------------------
// B-fragment swizzle (16-bit B 32x16, contiguous K order of this code):
//   lane l holds column N = nt*16 + (l&15); K-half = (l>>4)*16; dword v holds
//   K = 2v, 2v+1 within that half.  Stored as [kc][nt][lane][8 dwords].
// ---------------------------------------------------------------------------
__global__ void wib_swizzle_kernel(const float* __restrict__ Wi,
                                   unsigned int* __restrict__ Wib) {
  int tid = blockIdx.x * blockDim.x + threadIdx.x;       // ((kc*NT+nt)*32+lane)*8+v
  if (tid >= KC_X * NT * 32 * 8) return;
  int v    = tid & 7;
  int lane = (tid >> 3) & 31;
  int g    = tid >> 8;                                   // kc*NT + nt
  int nt   = g % NT;
  int kc   = g / NT;
  int K = kc * 32 + ((lane >> 4) << 4) + 2 * v;
  int N = nt * 16 + (lane & 15);
  float lo = (K     < F_IN) ? Wi[(size_t)K * G3 + N]       : 0.0f;
  float hi = (K + 1 < F_IN) ? Wi[(size_t)(K + 1) * G3 + N] : 0.0f;
  Wib[tid] = packbf(lo, hi);
}

// Combined [Wh_rz | Wh_n] -> bf16 B-fragments. Columns 0..1023 = Wh_rz (r,z),
// 1024..1535 = Wh_n. K = 0..511 always in range.
__global__ void whb_swizzle_kernel(const float* __restrict__ Wh_rz,
                                   const float* __restrict__ Wh_n,
                                   unsigned int* __restrict__ Whb) {
  int tid = blockIdx.x * blockDim.x + threadIdx.x;
  if (tid >= KC_H * NT * 32 * 8) return;
  int v    = tid & 7;
  int lane = (tid >> 3) & 31;
  int g    = tid >> 8;
  int nt   = g % NT;
  int kc   = g / NT;
  int K = kc * 32 + ((v >> 2) << 4) + ((lane >> 4) << 3) + 2 * (v & 3);
  int N = nt * 16 + (lane & 15);
  float lo, hi;
  if (N < 2 * H) {
    lo = Wh_rz[(size_t)K * (2 * H) + N];
    hi = Wh_rz[(size_t)(K + 1) * (2 * H) + N];
  } else {
    lo = Wh_n[(size_t)K * H + (N - 2 * H)];
    hi = Wh_n[(size_t)(K + 1) * H + (N - 2 * H)];
  }
  Whb[tid] = packbf(lo, hi);
}

// ---------------------------------------------------------------------------
// Stage 2: xg[65536,1536] = X[65536,1030] @ Wi + bi.
// Block = one 16-row M tile, 8 waves x 12 N-tiles = all 96 N tiles.
// Per K-chunk: 256 threads cooperatively convert the 16x32 fp32 A tile to a
// 1 KB LDS buffer laid out exactly as the WMMA A fragment
//   (lane row = l&15, K = (l>>4)*16 + 2v),
// then each wave reads it with two ds_load_b128 and issues 12 WMMAs against
// L2-resident pre-swizzled B fragments.
// ---------------------------------------------------------------------------
__global__ void __launch_bounds__(256)
gemm_xw_kernel(const float* __restrict__ X,
               const unsigned int* __restrict__ Wib,
               const float* __restrict__ bi,
               float* __restrict__ xg) {
  __shared__ __align__(16) unsigned int a_lds[32 * 8];   // 1 KB staged A frag

  const int tid    = threadIdx.x;
  const int lane   = tid & 31;
  const int half   = lane >> 4;
  const int wv     = tid >> 5;          // wave 0..7
  const int m_tile = blockIdx.x;        // 0..4095
  const int n0t    = wv * 12;           // 12 N tiles per wave

  // cooperative fill mapping: thread -> (fragment lane fl, dword fv)
  const int fl = tid >> 3;              // 0..31
  const int fv = tid & 7;               // 0..7
  const float* xrow = X + (size_t)(m_tile * 16 + (fl & 15)) * F_IN;
  const int fkoff = ((fl >> 4) << 4) + (fv << 1);

  const uint4* wb = (const uint4*)Wib;
  const uint4* al = (const uint4*)a_lds;

  f32x8 c[12];
#pragma unroll
  for (int i = 0; i < 12; ++i) c[i] = f32x8{};

  for (int kc = 0; kc < KC_X; ++kc) {
    // ---- stage A chunk (each X element touched exactly once per block)
    const int K = kc * 32 + fkoff;
    unsigned int packed;
    if (K + 1 < F_IN) {                 // K even, 8B-aligned float2
      float2 e = *(const float2*)(xrow + K);
      packed = packbf(e.x, e.y);
    } else {
      float lo = (K < F_IN) ? xrow[K] : 0.0f;
      packed = packbf(lo, 0.0f);
    }
    __syncthreads();                    // prior chunk fully consumed
    a_lds[fl * 8 + fv] = packed;
    __syncthreads();                    // staged chunk visible

    BFrag a;
    a.q[0] = al[lane * 2];
    a.q[1] = al[lane * 2 + 1];

#pragma unroll
    for (int tt = 0; tt < 12; ++tt) {
      BFrag b;
      size_t idx = ((size_t)(kc * NT + n0t + tt) * 32 + lane) * 2;
      b.q[0] = wb[idx];
      b.q[1] = wb[idx + 1];
      c[tt] = wmma_bf16(a, b, c[tt]);
    }
  }

  // D layout: VGPR j -> row j + 8*half, lane -> col (lane&15). Add bias, store.
#pragma unroll
  for (int tt = 0; tt < 12; ++tt) {
    const int ncol = (n0t + tt) * 16 + (lane & 15);
    const float bias = bi[ncol];
#pragma unroll
    for (int j = 0; j < 8; ++j) {
      const int mrow = m_tile * 16 + j + half * 8;
      xg[(size_t)mrow * G3 + ncol] = c[tt][j] + bias;
    }
  }
}

// ---------------------------------------------------------------------------
// Stage 3: per-batch-slice GRU scan. Block b owns batch rows [16b,16b+16).
// 16 waves: wave w owns batch row w for elementwise (h in registers, fp32),
// all waves cooperate on the [16,512]@[512,512] gate GEMMs (bf16 WMMA),
// A-fragments streamed from LDS with ds_load_b128, B from L2-hot Whb.
// Phases per step: p=0 -> r, p=1 -> z, p=2 -> n + state update.
// ---------------------------------------------------------------------------
__global__ void __launch_bounds__(512)
gru_scan_kernel(const float* __restrict__ h0,
                const float* __restrict__ xg,
                const unsigned int* __restrict__ Whb,
                const float* __restrict__ bh_n,
                float* __restrict__ carry_out,
                float* __restrict__ ys) {
  __shared__ __align__(16) unsigned short h_bf[16 * H];   // 16 KB bf16 hidden
  __shared__ float s_buf[16 * H];                         // 32 KB gate preacts

  const int w    = threadIdx.x >> 5;    // wave id == local batch row (0..15)
  const int lane = threadIdx.x & 31;
  const int half = lane >> 4;
  const int brow = blockIdx.x * 16 + w; // global batch row
  const int arow = lane & 15;           // WMMA A-fragment row

  float h[16], bh[16], rr[16], zz[16];
#pragma unroll
  for (int i = 0; i < 16; ++i) {
    const int col = lane + 32 * i;
    h[i]  = h0[(size_t)brow * H + col];
    bh[i] = bh_n[col];
    h_bf[w * H + col] = f2bfu(h[i]);
  }
  __syncthreads();

  const uint4* hb4 = (const uint4*)h_bf;
  const uint4* wb  = (const uint4*)Whb;

  for (int t = 0; t < T_STEPS; ++t) {
    const size_t xbase = ((size_t)t * BATCH + brow) * G3;

    for (int p = 0; p < 3; ++p) {
      // ---- GEMM phase: s = h @ W_gate(p), wave w computes cols [32w,32w+32)
      f32x8 c0 = {}, c1 = {};
      const int nt0 = p * 32 + w * 2;
#pragma unroll
      for (int kc = 0; kc < KC_H; ++kc) {
        BFrag a;
        const int aidx = arow * 64 + kc * 4 + half;  // uint4 units into h_bf
        a.q[0] = hb4[aidx];
        a.q[1] = hb4[aidx + 2];
        BFrag b0, b1;
        size_t i0 = ((size_t)(kc * NT + nt0) * 32 + lane) * 2;
        size_t i1 = i0 + 64;                         // next N tile
        b0.q[0] = wb[i0];  b0.q[1] = wb[i0 + 1];
        b1.q[0] = wb[i1];  b1.q[1] = wb[i1 + 1];
        c0 = wmma_bf16(a, b0, c0);
        c1 = wmma_bf16(a, b1, c1);
      }
#pragma unroll
      for (int j = 0; j < 8; ++j) {
        const int row = j + half * 8;
        s_buf[row * H + w * 32 + (lane & 15)]      = c0[j];
        s_buf[row * H + w * 32 + 16 + (lane & 15)] = c1[j];
      }
      __syncthreads();

      // ---- elementwise consume (wave w == batch row w, lane owns col l+32i)
      if (p == 0) {
#pragma unroll
        for (int i = 0; i < 16; ++i) {
          const int col = lane + 32 * i;
          const float pre = xg[xbase + col] + s_buf[w * H + col];
          rr[i] = pre / (1.0f + __expf(-pre));               // silu
        }
      } else if (p == 1) {
#pragma unroll
        for (int i = 0; i < 16; ++i) {
          const int col = lane + 32 * i;
          const float pre = xg[xbase + H + col] + s_buf[w * H + col];
          zz[i] = pre / (1.0f + __expf(-pre));               // silu
        }
      } else {
#pragma unroll
        for (int i = 0; i < 16; ++i) {
          const int col = lane + 32 * i;
          const float hn  = s_buf[w * H + col] + bh[i];
          const float pre = xg[xbase + 2 * H + col] + rr[i] * hn;
          const float n   = tanhf(pre);
          const float hv  = (1.0f - zz[i]) * n + zz[i] * h[i];
          h[i] = hv;
          ys[((size_t)t * BATCH + brow) * H + col] = hv;
          h_bf[w * H + col] = f2bfu(hv);
        }
      }
      __syncthreads();
    }
  }

#pragma unroll
  for (int i = 0; i < 16; ++i) {
    const int col = lane + 32 * i;
    carry_out[(size_t)brow * H + col] = h[i];
  }
}

// ---------------------------------------------------------------------------
extern "C" void kernel_launch(void* const* d_in, const int* in_sizes, int n_in,
                              void* d_out, int out_size, void* d_ws, size_t ws_size,
                              hipStream_t stream) {
  (void)in_sizes; (void)n_in; (void)out_size; (void)ws_size;

  const float* h0    = (const float*)d_in[0];   // [128,512]
  const float* X     = (const float*)d_in[1];   // [512,128,1030]
  const float* Wi    = (const float*)d_in[2];   // [1030,1536]
  const float* bi    = (const float*)d_in[3];   // [1536]
  const float* Wh_rz = (const float*)d_in[4];   // [512,1024]
  const float* Wh_n  = (const float*)d_in[5];   // [512,512]
  const float* bh_n  = (const float*)d_in[6];   // [512]

  float* out   = (float*)d_out;
  float* carry = out;                    // [128,512]
  float* ys    = out + BATCH * H;        // [512,128,512]

  char* ws = (char*)d_ws;                // needs ~408 MB
  float*        xg  = (float*)ws;
  unsigned int* Wib = (unsigned int*)(ws + XG_BYTES);
  unsigned int* Whb = (unsigned int*)(ws + XG_BYTES + WIB_BYTES);

  wib_swizzle_kernel<<<(KC_X * NT * 32 * 8 + 255) / 256, 256, 0, stream>>>(Wi, Wib);
  whb_swizzle_kernel<<<(KC_H * NT * 32 * 8 + 255) / 256, 256, 0, stream>>>(Wh_rz, Wh_n, Whb);
  gemm_xw_kernel<<<M_TILES, 256, 0, stream>>>(X, Wib, bi, xg);
  gru_scan_kernel<<<8, 512, 0, stream>>>(h0, xg, Whb, bh_n, carry, ys);
}
